// GNN_35459249996394
// MI455X (gfx1250) — matrix operand order repaired and checked by reading, not compiled, not run on hardware
//
#include <hip/hip_runtime.h>
#include <hip/hip_bf16.h>
#include <math.h>

typedef __attribute__((ext_vector_type(16))) _Float16 v16h;
typedef __attribute__((ext_vector_type(8)))  _Float16 v8h;
typedef __attribute__((ext_vector_type(8)))  float    v8f;

#define NNODES 50000
#define NEDGES 640000
#define NGRAPH 64
#define DH 128

// ---------------------------------------------------------------- utilities
__global__ void __launch_bounds__(256) fill_f32_kernel(float* p, float v, int n) {
    int i = blockIdx.x * 256 + threadIdx.x;
    if (i < n) p[i] = v;
}
__global__ void __launch_bounds__(256) fill_u32_kernel(unsigned* p, unsigned v, int n) {
    int i = blockIdx.x * 256 + threadIdx.x;
    if (i < n) p[i] = v;
}

// ------------------------------------------------------------- degree/dinv
__global__ void __launch_bounds__(256) deg_kernel(const int* __restrict__ dst,
                                                  float* __restrict__ deg, int E) {
    int e = blockIdx.x * 256 + threadIdx.x;
    if (e < E) atomicAdd(&deg[dst[e]], 1.0f);
}
__global__ void __launch_bounds__(256) dinv_kernel(float* deg, int n) {
    int i = blockIdx.x * 256 + threadIdx.x;
    if (i < n) deg[i] = rsqrtf(deg[i]);   // deg >= 1 (self loop)
}

// ---------------------------------------------------------------- encoder
__global__ void __launch_bounds__(256) encoder_kernel(const float* __restrict__ x,
                                                      const float* __restrict__ ew,
                                                      const float* __restrict__ eb,
                                                      _Float16* __restrict__ h16, int n) {
    int idx = blockIdx.x * 256 + threadIdx.x;
    if (idx >= n * DH) return;
    int node = idx >> 7, d = idx & (DH - 1);
    float v = eb[d] + x[node * 3 + 0] * ew[0 * DH + d]
                    + x[node * 3 + 1] * ew[1 * DH + d]
                    + x[node * 3 + 2] * ew[2 * DH + d];
    v = v > 0.0f ? v : (expf(v) - 1.0f);          // ELU
    h16[idx] = (_Float16)v;
}

// ---------------------------------------------------------------------------
// Weight f32 -> f16, permuted into WMMA B-fragment order (+ zero LN reduction).
// Fragment layout: idx = ((wave*4 + s)*32 + lane)*16 + i
//   value = W[k][n],  k = s*32 + (lane>>4)*16 + i,  n = (lane&15) + 16*wave
// Each lane's 16-half fragment is then one contiguous 32-byte run.
// ---------------------------------------------------------------------------
__global__ void __launch_bounds__(256) convert_w_kernel(const float* __restrict__ w,
                                                        _Float16* __restrict__ wfrag,
                                                        float* __restrict__ red) {
    int idx = blockIdx.x * 256 + threadIdx.x;
    if (idx < DH * DH) {
        int wv   = idx >> 11;          // wave  (0..7)  -> N tile
        int s    = (idx >> 9) & 3;     // K step (0..3)
        int lane = (idx >> 4) & 31;    // lane  (0..31)
        int i    = idx & 15;           // element within fragment
        int k = s * 32 + (lane >> 4) * 16 + i;
        int n = (lane & 15) + 16 * wv;
        wfrag[idx] = (_Float16)w[k * DH + n];
    }
    if (blockIdx.x == 0 && threadIdx.x < 4) red[threadIdx.x] = 0.0f;
}

// ---------------------------------------------------------------------------
// WMMA GEMM: hW = h16 @ W.  No LDS: A rows and pre-permuted B fragments are
// read straight from global (both L2-resident).  Block = 256 thr = 8 wave32,
// each wave owns one 16x16 tile of a 16x128 output strip.
// ---------------------------------------------------------------------------
__global__ void __launch_bounds__(256) gemm_wmma_kernel(const _Float16* __restrict__ A,
                                                        const _Float16* __restrict__ Bf,
                                                        float* __restrict__ C) {
    const int tid  = threadIdx.x;
    const int wave = tid >> 5;                 // 0..7 -> N tile
    const int lane = tid & 31;
    const int rowBase = blockIdx.x * 16;
    const int m   = lane & 15;
    const int grp = lane >> 4;                 // 0/1 half-lane group
    const int n   = (lane & 15) + wave * 16;   // output column

    const _Float16* arow = A + (size_t)(rowBase + m) * DH;
    v8f acc = {};
    #pragma unroll
    for (int s = 0; s < 4; ++s) {              // K = 128 = 4 x 32
        // A fragment: two 8-half (16B) runs covering k = grp*16 + 0..15 (matches B)
        v8h alo = *(const v8h*)(arow + s * 32 + grp * 16);
        v8h ahi = *(const v8h*)(arow + s * 32 + grp * 16 + 8);
        v16h a = __builtin_shufflevector(alo, ahi, 0, 1, 2, 3, 4, 5, 6, 7,
                                                   8, 9, 10, 11, 12, 13, 14, 15);
        // B fragment: one contiguous 32B run from the pre-permuted weight
        v16h b = *(const v16h*)(Bf + (((wave * 4 + s) * 32 + lane) << 4));
        acc = __builtin_amdgcn_wmma_f32_16x16x32_f16(false, a, false, b,
                                                     (short)0, acc, false, false);
    }
    // D layout: VGPR r -> row grp*8 + r, col n
    #pragma unroll
    for (int r = 0; r < 8; ++r)
        C[(size_t)(rowBase + grp * 8 + r) * DH + n] = acc[r];
}

// ----------------------------------------------------- edge gather/scale/scatter-add
// Two edges per 256-thread block; 128 lanes = one coalesced feature row per edge.
__global__ void __launch_bounds__(256) propagate_kernel(const float* __restrict__ hW,
                                                        const int* __restrict__ src,
                                                        const int* __restrict__ dst,
                                                        const float* __restrict__ dinv,
                                                        float* __restrict__ aggr,
                                                        int E, int n) {
    int e = blockIdx.x * 2 + (threadIdx.x >> 7);
    int d = threadIdx.x & (DH - 1);
    if (e >= E + n) return;
    int s, t;
    if (e < E) { s = src[e]; t = dst[e]; } else { s = t = e - E; }  // self loops
    float w = dinv[s] * dinv[t];
    atomicAdd(&aggr[(size_t)t * DH + d], hW[(size_t)s * DH + d] * w);
}

// ------------------------------------------- bias + ReLU + global sum/sumsq reduce
__global__ void __launch_bounds__(256) bias_relu_reduce_kernel(float* __restrict__ aggr,
                                                               const float* __restrict__ b,
                                                               float* __restrict__ red,
                                                               int total) {
    __shared__ float ssum[256], ssq[256];
    int idx = blockIdx.x * 256 + threadIdx.x;
    float v = 0.0f;
    if (idx < total) {
        v = aggr[idx] + b[idx & (DH - 1)];
        v = v > 0.0f ? v : 0.0f;
        aggr[idx] = v;
    }
    ssum[threadIdx.x] = v;
    ssq[threadIdx.x]  = v * v;
    __syncthreads();
    for (int s = 128; s > 0; s >>= 1) {
        if (threadIdx.x < s) {
            ssum[threadIdx.x] += ssum[threadIdx.x + s];
            ssq[threadIdx.x]  += ssq[threadIdx.x + s];
        }
        __syncthreads();
    }
    if (threadIdx.x == 0) {
        atomicAdd(&red[0], ssum[0]);
        atomicAdd(&red[1], ssq[0]);
    }
}

// ---------------------------------------------- graph-LN apply (+emit f16 for next GEMM)
__global__ void __launch_bounds__(256) ln_apply_kernel(float* __restrict__ aggr,
                                                       _Float16* __restrict__ h16,
                                                       const float* __restrict__ lw,
                                                       const float* __restrict__ lb,
                                                       const float* __restrict__ red,
                                                       int total) {
    int idx = blockIdx.x * 256 + threadIdx.x;
    if (idx >= total) return;
    float inv_n = 1.0f / (float)total;
    float mu  = red[0] * inv_n;
    float var = fmaxf(red[1] * inv_n - mu * mu, 0.0f);
    float sd  = sqrtf(var);
    int d = idx & (DH - 1);
    float v = (aggr[idx] - mu) / (sd + 1e-5f) * lw[d] + lb[d];
    aggr[idx] = v;
    h16[idx] = (_Float16)v;
}

// -------------------------------------------------------------- segment max pool
__device__ __forceinline__ unsigned enc_ord(float f) {
    unsigned u = __float_as_uint(f);
    return (u & 0x80000000u) ? ~u : (u | 0x80000000u);
}
__global__ void __launch_bounds__(256) pool_max_kernel(const float* __restrict__ h,
                                                       const int* __restrict__ batch,
                                                       unsigned* __restrict__ gmax, int n) {
    int idx = blockIdx.x * 256 + threadIdx.x;
    if (idx >= n * DH) return;
    int node = idx >> 7, d = idx & (DH - 1);
    atomicMax(&gmax[batch[node] * DH + d], enc_ord(h[idx]));
}

// ------------------------------------------------------------------- MLP head
__global__ void __launch_bounds__(256) head_kernel(const unsigned* __restrict__ gmax,
                                                   const float* __restrict__ fc_w,
                                                   const float* __restrict__ fc_b,
                                                   const float* __restrict__ pred_w,
                                                   const float* __restrict__ pred_b,
                                                   float* __restrict__ tbuf,
                                                   float* __restrict__ out) {
    __shared__ float g[NGRAPH * DH];                      // 32 KB
    int tid = threadIdx.x;
    for (int i = tid; i < NGRAPH * DH; i += 256) {
        unsigned key = gmax[i];
        unsigned u = (key & 0x80000000u) ? (key ^ 0x80000000u) : ~key;
        g[i] = __uint_as_float(u);
    }
    __syncthreads();
    for (int o = tid; o < NGRAPH * DH; o += 256) {        // t = tanh(g@fc_w + fc_b)
        int i = o >> 7, j = o & (DH - 1);
        float acc = fc_b[j];
        #pragma unroll 4
        for (int k = 0; k < DH; ++k) acc += g[i * DH + k] * fc_w[k * DH + j];
        tbuf[o] = tanhf(acc);
    }
    __threadfence();
    __syncthreads();
    for (int o = tid; o < NGRAPH * 10; o += 256) {        // out = t@pred_w + pred_b
        int i = o / 10, c = o % 10;
        float acc = pred_b[c];
        #pragma unroll 4
        for (int k = 0; k < DH; ++k) acc += tbuf[i * DH + k] * pred_w[k * 10 + c];
        out[i * 10 + c] = acc;
    }
}

// ---------------------------------------------------------------------- launch
extern "C" void kernel_launch(void* const* d_in, const int* in_sizes, int n_in,
                              void* d_out, int out_size, void* d_ws, size_t ws_size,
                              hipStream_t stream) {
    const float* x      = (const float*)d_in[0];
    const int*   src    = (const int*)d_in[1];
    const int*   dst    = (const int*)d_in[2];
    const int*   batch  = (const int*)d_in[3];
    const float* enc_w  = (const float*)d_in[4];
    const float* enc_b  = (const float*)d_in[5];
    const float* fc_w   = (const float*)d_in[6];
    const float* fc_b   = (const float*)d_in[7];
    const float* pred_w = (const float*)d_in[8];
    const float* pred_b = (const float*)d_in[9];

    const int N = NNODES, E = NEDGES;
    const int total = N * DH;          // 6,400,000

    char* ws = (char*)d_ws;
    _Float16* h16   = (_Float16*)ws; ws += (size_t)N * DH * 2;     // 12.8 MB
    float*    hW    = (float*)ws;    ws += (size_t)N * DH * 4;     // 25.6 MB
    float*    aggr  = (float*)ws;    ws += (size_t)N * DH * 4;     // 25.6 MB
    _Float16* wfrag = (_Float16*)ws; ws += DH * DH * 2;            // 32 KB
    float*    dinv  = (float*)ws;    ws += ((size_t)N * 4 + 255) / 256 * 256;
    unsigned* gmax  = (unsigned*)ws; ws += NGRAPH * DH * 4;        // 32 KB
    float*    tbuf  = (float*)ws;    ws += NGRAPH * DH * 4;        // 32 KB
    float*    red   = (float*)ws;    ws += 256;

    // degrees (self-loop => init to 1), then dinv = rsqrt(deg)
    fill_f32_kernel<<<(N + 255) / 256, 256, 0, stream>>>(dinv, 1.0f, N);
    deg_kernel<<<(E + 255) / 256, 256, 0, stream>>>(dst, dinv, E);
    dinv_kernel<<<(N + 255) / 256, 256, 0, stream>>>(dinv, N);

    // encoder -> h16
    encoder_kernel<<<(total + 255) / 256, 256, 0, stream>>>(x, enc_w, enc_b, h16, N);

    for (int l = 0; l < 4; ++l) {
        const float* w  = (const float*)d_in[10 + 4 * l];
        const float* b  = (const float*)d_in[11 + 4 * l];
        const float* lw = (const float*)d_in[12 + 4 * l];
        const float* lb = (const float*)d_in[13 + 4 * l];
        convert_w_kernel<<<64, 256, 0, stream>>>(w, wfrag, red);
        gemm_wmma_kernel<<<N / 16, 256, 0, stream>>>(h16, wfrag, hW);
        fill_f32_kernel<<<(total + 255) / 256, 256, 0, stream>>>(aggr, 0.0f, total);
        propagate_kernel<<<(E + N + 1) / 2, 256, 0, stream>>>(hW, src, dst, dinv, aggr, E, N);
        bias_relu_reduce_kernel<<<total / 256, 256, 0, stream>>>(aggr, b, red, total);
        ln_apply_kernel<<<total / 256, 256, 0, stream>>>(aggr, h16, lw, lb, red, total);
    }

    fill_u32_kernel<<<(NGRAPH * DH + 255) / 256, 256, 0, stream>>>(gmax, 0u, NGRAPH * DH);
    pool_max_kernel<<<(total + 255) / 256, 256, 0, stream>>>(aggr, batch, gmax, N);
    head_kernel<<<1, 256, 0, stream>>>(gmax, fc_w, fc_b, pred_w, pred_b, tbuf, (float*)d_out);
}
